// ResonantManifold_55095840473096
// MI455X (gfx1250) — matrix operand order repaired and checked by reading, not compiled, not run on hardware
//
#include <hip/hip_runtime.h>

typedef __attribute__((ext_vector_type(16))) _Float16 v16h;
typedef __attribute__((ext_vector_type(8)))  float    v8f;

#define D     256
#define CELLS 8
#define BATCH 2
#define SEQ   2048
#define BC    (BATCH*CELLS)      // 16
#define ROWS  (BC*SEQ)           // 32768

// padded LDS strides (halfs) so 16-lane gathers hit 16 distinct banks
#define KVS 264                  // key/value tile row stride (row-major [key][dim])
#define VTS 34                   // V^T tile row stride       ([dim][key])
#define PLS 34                   // P scratch row stride      ([qrow][key])

union AF { v16h h; unsigned int u[8]; };

// ---------------------------------------------------------------------------
// Kernel 1: gate (DxD f32, row-major [k][n]) -> gateT f16 [n][k]
// (so WMMA B-fragment k-pairs are contiguous 32-bit loads)
// ---------------------------------------------------------------------------
__global__ void rm_gateT(const float* __restrict__ gate, _Float16* __restrict__ gateT) {
    int k = blockIdx.x;          // 0..255
    int n = threadIdx.x;         // 0..255
    gateT[n * D + k] = (_Float16)gate[k * D + n];
}

// ---------------------------------------------------------------------------
// Kernel 2: per-row LayerNorm -> nx f16  (one 256-thread block per row)
// ---------------------------------------------------------------------------
__global__ void rm_layernorm(const float* __restrict__ x, _Float16* __restrict__ nxh) {
    __shared__ float s1[256];
    __shared__ float s2[256];
    int row = blockIdx.x;
    int t   = threadIdx.x;
    float v = x[(size_t)row * D + t];
    s1[t] = v;
    s2[t] = v * v;
    __syncthreads();
    for (int off = 128; off > 0; off >>= 1) {
        if (t < off) { s1[t] += s1[t + off]; s2[t] += s2[t + off]; }
        __syncthreads();
    }
    float mu  = s1[0] * (1.0f / D);
    float var = s2[0] * (1.0f / D) - mu * mu;
    float r   = rsqrtf(var + 1e-5f);
    nxh[(size_t)row * D + t] = (_Float16)((v - mu) * r);
}

// ---------------------------------------------------------------------------
// Kernel 3: qkv = nx @ gate  via v_wmma_f32_16x16x32_f16, output as f16.
// One 16x16 output tile per wave; 4 waves (128 threads) per block.
// ---------------------------------------------------------------------------
__global__ void __launch_bounds__(128) rm_qkv(const _Float16* __restrict__ nxh,
                                              const _Float16* __restrict__ gateT,
                                              _Float16* __restrict__ qkvh) {
    int wave = threadIdx.x >> 5;
    int lane = threadIdx.x & 31;
    int tile = blockIdx.x * 4 + wave;       // 32768 tiles total
    int tm = tile >> 4;                     // 0..2047  (M tiles)
    int tn = tile & 15;                     // 0..15    (N tiles)
    int mBase = tm * 16, nBase = tn * 16;

    const unsigned int* nxu = (const unsigned int*)nxh;
    const unsigned int* gtu = (const unsigned int*)gateT;
    int arow = mBase + (lane & 15);
    int bcol = nBase + (lane & 15);

    v8f acc = {};
    for (int kc = 0; kc < 8; ++kc) {
        AF a, b;
        int k0 = kc * 32;
#pragma unroll
        for (int i = 0; i < 8; ++i) {
            // A layout (16-bit, 16x32): k = 16*(i/4) + 8*(lane/16) + 2*(i%4)
            int ka = k0 + ((i >> 2) << 4) + ((lane >> 4) << 3) + ((i & 3) << 1);
            a.u[i] = nxu[((size_t)arow * D + ka) >> 1];
            // B layout (16-bit, 32x16): k = 16*(i/4) + 8*(lane/16) + 2*(i%4) ; col = lane%16
            int kb = k0 + ((i >> 2) << 4) + ((lane >> 4) << 3) + ((i & 3) << 1);
            b.u[i] = gtu[((size_t)bcol * D + kb) >> 1];
        }
        acc = __builtin_amdgcn_wmma_f32_16x16x32_f16(false, a.h, false, b.h,
                                                     (short)0, acc, false, false);
    }
#pragma unroll
    for (int v = 0; v < 8; ++v) {
        int r = mBase + v + ((lane >> 4) << 3);
        int c = nBase + (lane & 15);
        qkvh[(size_t)r * D + c] = (_Float16)acc[v];
    }
}

// ---------------------------------------------------------------------------
// Kernel 4: causal flash attention + residual.  xa = x + softmax(QK^T/16) V
// One wave owns a 16-row tile (full D=256 output in registers).
// Key blocks of 32 staged into LDS with GLOBAL_LOAD_ASYNC_TO_LDS_B128
// (ASYNCcnt), then consumed by WMMA via LDS gathers.
// ---------------------------------------------------------------------------
__global__ void __launch_bounds__(128) rm_attn(const float* __restrict__ x,
                                               const _Float16* __restrict__ qkvh,
                                               float* __restrict__ xa) {
    __shared__ _Float16 kv[32 * KVS];        // K/V block row-major [key][dim] (~16.5 KB)
    __shared__ _Float16 vt[D * VTS];         // V^T tile [dim][key]            (~17 KB)
    __shared__ _Float16 plds[4 * 16 * PLS];  // per-wave P scratch             (~4.3 KB)

    int wave = threadIdx.x >> 5;
    int lane = threadIdx.x & 31;
    int bc   = blockIdx.y;
    int rt0  = blockIdx.x * 4;
    int rt   = rt0 + wave;
    int rowBase = rt * 16;
    const size_t base = (size_t)bc * SEQ * D;
    const unsigned int* qk_u = (const unsigned int*)(qkvh + base);

    // cache Q fragments for all 8 k-chunks (A layout)
    AF q[8];
    int qrow = rowBase + (lane & 15);
#pragma unroll
    for (int kc = 0; kc < 8; ++kc)
#pragma unroll
        for (int i = 0; i < 8; ++i) {
            int ka = kc * 32 + ((i >> 2) << 4) + ((lane >> 4) << 3) + ((i & 3) << 1);
            q[kc].u[i] = qk_u[((size_t)qrow * D + ka) >> 1];
        }

    float mrow[8], lrow[8];
    v8f o[16] = {};
#pragma unroll
    for (int v = 0; v < 8; ++v) { mrow[v] = -1e30f; lrow[v] = 0.0f; }

    int myJmax  = (rowBase + 15) >> 5;
    int blkJmax = ((rt0 + 3) * 16 + 15) >> 5;
    _Float16* pw = plds + wave * 16 * PLS;
    const float scale = 0.0625f;             // D^-0.5
    const unsigned kv_base  = (unsigned)(uintptr_t)&kv[0];

    for (int j = 0; j <= blkJmax; ++j) {
        int keyBase = j * 32;

        // ---- async stage K/V block (32 x 256 f16) global -> LDS, row-major
        __syncthreads();                     // previous-iteration readers done
        for (int idx = threadIdx.x; idx < 32 * 32; idx += 128) {
            int r  = idx >> 5;               // key row 0..31
            int c8 = (idx & 31) << 3;        // dim chunk (8 halfs = 16 B)
            unsigned loff = kv_base + (unsigned)(r * KVS + c8) * 2u;
            unsigned long long gaddr =
                (unsigned long long)(uintptr_t)(qkvh + base + (size_t)(keyBase + r) * D + c8);
            asm volatile("global_load_async_to_lds_b128 %0, %1, off"
                         :: "v"(loff), "v"(gaddr) : "memory");
        }
        asm volatile("s_wait_asynccnt 0x0" ::: "memory");
        __syncthreads();                     // whole block's async stores visible

        // ---- build V^T tile (LDS -> LDS transpose, padded strides)
        for (int idx = threadIdx.x; idx < 32 * 32; idx += 128) {
            int r  = idx >> 5;
            int c8 = (idx & 31) << 3;
            union { uint4 u4; _Float16 e[8]; } cv;
            cv.u4 = *(const uint4*)&kv[r * KVS + c8];
#pragma unroll
            for (int e = 0; e < 8; ++e) vt[(c8 + e) * VTS + r] = cv.e[e];
        }
        __syncthreads();
        if (j > myJmax) continue;            // uniform per wave

        // ---- S = Q K^T  (16x32 tile = two 16x16 WMMA chains over K=256)
        v8f s0 = {}, s1 = {};
        const unsigned int* kvu = (const unsigned int*)kv;
        int krl = lane & 15;                 // local key row (cols 0-15)
#pragma unroll
        for (int kc = 0; kc < 8; ++kc) {
            AF b0, b1;
#pragma unroll
            for (int i = 0; i < 8; ++i) {
                int kb = kc * 32 + ((i >> 2) << 4) + ((lane >> 4) << 3) + ((i & 3) << 1);
                b0.u[i] = kvu[(krl * KVS + kb) >> 1];
                b1.u[i] = kvu[((krl + 16) * KVS + kb) >> 1];
            }
            s0 = __builtin_amdgcn_wmma_f32_16x16x32_f16(false, q[kc].h, false, b0.h,
                                                        (short)0, s0, false, false);
            s1 = __builtin_amdgcn_wmma_f32_16x16x32_f16(false, q[kc].h, false, b1.h,
                                                        (short)0, s1, false, false);
        }

        // ---- online softmax (rows live in VGPR index, cols in 16 lanes)
        bool diag = (j == myJmax);
#pragma unroll
        for (int v = 0; v < 8; ++v) {
            float a0 = s0[v] * scale, a1 = s1[v] * scale;
            int grow = rowBase + v + ((lane >> 4) << 3);
            if (diag) {
                int gc0 = keyBase + (lane & 15);
                if (gc0 > grow)      a0 = -1e30f;
                if (gc0 + 16 > grow) a1 = -1e30f;
            }
            float mx = fmaxf(a0, a1);
#pragma unroll
            for (int w = 1; w < 16; w <<= 1) mx = fmaxf(mx, __shfl_xor(mx, w, 32));
            float mnew = fmaxf(mrow[v], mx);
            float corr = __expf(mrow[v] - mnew);
            float p0 = __expf(a0 - mnew);
            float p1 = __expf(a1 - mnew);
            float sum = p0 + p1;
#pragma unroll
            for (int w = 1; w < 16; w <<= 1) sum += __shfl_xor(sum, w, 32);
            lrow[v] = lrow[v] * corr + sum;
            mrow[v] = mnew;
#pragma unroll
            for (int t = 0; t < 16; ++t) o[t][v] *= corr;
            // stash P (C layout) into wave-private LDS for A-layout re-read
            int prow = v + ((lane >> 4) << 3);
            int pc   = lane & 15;
            pw[prow * PLS + pc]      = (_Float16)p0;
            pw[prow * PLS + pc + 16] = (_Float16)p1;
        }

        // wave-local LDS handoff: wait for our ds stores before cross-lane reads
        asm volatile("s_wait_dscnt 0x0" ::: "memory");

        AF p;
        {
            const unsigned int* pu = (const unsigned int*)pw;
            int prow = lane & 15;
#pragma unroll
            for (int i = 0; i < 8; ++i) {
                int kp = ((i >> 2) << 4) + ((lane >> 4) << 3) + ((i & 3) << 1);
                p.u[i] = pu[(prow * PLS + kp) >> 1];
            }
        }

        // ---- O += P V   (16 column tiles of 16, K = 32 keys)
        const unsigned int* vtu = (const unsigned int*)vt;
#pragma unroll
        for (int t = 0; t < 16; ++t) {
            AF bv;
            int n = t * 16 + (lane & 15);
#pragma unroll
            for (int i = 0; i < 8; ++i) {
                int kvk = ((i >> 2) << 4) + ((lane >> 4) << 3) + ((i & 3) << 1);
                bv.u[i] = vtu[(n * VTS + kvk) >> 1];
            }
            o[t] = __builtin_amdgcn_wmma_f32_16x16x32_f16(false, p.h, false, bv.h,
                                                          (short)0, o[t], false, false);
        }
    }

    // ---- finalize: xa = x + O / l   (residual fused)
#pragma unroll
    for (int t = 0; t < 16; ++t)
#pragma unroll
        for (int v = 0; v < 8; ++v) {
            int r = rowBase + v + ((lane >> 4) << 3);
            int c = t * 16 + (lane & 15);
            size_t idx = base + (size_t)r * D + c;
            xa[idx] = x[idx] + o[t][v] / lrow[v];
        }
}

// ---------------------------------------------------------------------------
// Kernel 5: cross-cell mix (8x8 inhibit), tanh, sin pulse -> out
// ---------------------------------------------------------------------------
__global__ void rm_mix(const float* __restrict__ xa, const float* __restrict__ inhibit,
                       const float* __restrict__ phases, const float* __restrict__ ambition,
                       float* __restrict__ out) {
    __shared__ float inh[64];
    __shared__ float ph[8];
    __shared__ float am[8];
    int t = threadIdx.x;
    if (t < 64) inh[t] = inhibit[t];
    if (t < 8) { ph[t] = phases[t]; am[t] = ambition[t]; }
    __syncthreads();

    size_t e  = (size_t)blockIdx.x * 256 + t;   // over BATCH*SEQ*D
    int    b  = (int)(e / ((size_t)SEQ * D));
    size_t td = e % ((size_t)SEQ * D);

    float xs[8];
#pragma unroll
    for (int c = 0; c < 8; ++c)
        xs[c] = xa[(size_t)(b * CELLS + c) * SEQ * D + td];
#pragma unroll
    for (int k = 0; k < 8; ++k) {
        float comp = 0.0f;
#pragma unroll
        for (int c = 0; c < 8; ++c) comp += xs[c] * inh[c * 8 + k];
        float xv    = xs[k] + tanhf(comp);
        float pulse = __sinf(xv * am[k] + ph[k]);
        out[(size_t)(b * CELLS + k) * SEQ * D + td] = xv + pulse * 0.02f;
    }
}

// ---------------------------------------------------------------------------
extern "C" void kernel_launch(void* const* d_in, const int* in_sizes, int n_in,
                              void* d_out, int out_size, void* d_ws, size_t ws_size,
                              hipStream_t stream) {
    (void)in_sizes; (void)n_in; (void)out_size; (void)ws_size;
    const float* x        = (const float*)d_in[0];
    /* d_in[1] = mask: causal by construction, hardcoded in rm_attn */
    const float* gate     = (const float*)d_in[2];
    const float* inhibit  = (const float*)d_in[3];
    const float* phases   = (const float*)d_in[4];
    const float* ambition = (const float*)d_in[5];
    float* out = (float*)d_out;

    char* ws = (char*)d_ws;
    _Float16* nxh   = (_Float16*)ws;                                   // 16 MiB
    _Float16* qkvh  = (_Float16*)(ws + (size_t)ROWS * D * 2);          // 16 MiB
    _Float16* gateT = (_Float16*)(ws + (size_t)ROWS * D * 4);          // 128 KiB
    float*    xa    = (float*)(ws + (size_t)ROWS * D * 4 + (size_t)D * D * 2); // 32 MiB

    rm_gateT    <<<dim3(D),    dim3(D),   0, stream>>>(gate, gateT);
    rm_layernorm<<<dim3(ROWS), dim3(D),   0, stream>>>(x, nxh);
    rm_qkv      <<<dim3((ROWS / 16) * (D / 16) / 4), dim3(128), 0, stream>>>(nxh, gateT, qkvh);
    rm_attn     <<<dim3(SEQ / 64, BC), dim3(128), 0, stream>>>(x, qkvh, xa);
    rm_mix      <<<dim3(BATCH * SEQ * D / 256), dim3(256), 0, stream>>>(
                    xa, inhibit, phases, ambition, out);
}
